// AttentionHead_44152263803147
// MI455X (gfx1250) — matrix operand order repaired and checked by reading, not compiled, not run on hardware
//
#include <hip/hip_runtime.h>

typedef __attribute__((ext_vector_type(16))) __bf16 v16bf;
typedef __attribute__((ext_vector_type(8)))  __bf16 v8bf;
typedef __attribute__((ext_vector_type(8)))  float  v8f;

#define NEG_INF (-3.402823466e+38f)

constexpr int N_CTX   = 4096;
constexpr int D_MODEL = 2048;

constexpr int BM = 128;        // block rows
constexpr int BN = 128;        // block cols
constexpr int BK = 32;         // k per wmma step (bf16 16x16x32)
constexpr int LDS_A = BK + 8;  // 40 bf16 = 80 B rows: 16B-aligned + bank-conflict pad
constexpr int LDS_B = BK + 8;
constexpr int NBUF  = 4;       // power-of-2 ring: buf = ki & 3 (no div in hot loop)

// ---------------------------------------------------------------------------
// gfx1250 async copy: 16 bytes global -> LDS per lane (ASYNCcnt tracked)
// ---------------------------------------------------------------------------
__device__ __forceinline__ void async_copy16(__bf16* lds_dst, const __bf16* gsrc)
{
    // Low 32 bits of a flat LDS pointer are the LDS byte offset (ISA 10.2).
    unsigned lds_off = (unsigned)(unsigned long long)lds_dst;
    asm volatile("global_load_async_to_lds_b128 %0, %1, off"
                 :: "v"(lds_off), "v"(gsrc) : "memory");
}
// Tile i complete while tile i+1's 4 per-thread ops remain in flight.
__device__ __forceinline__ void wait_async_le4()
{
    asm volatile("s_wait_asynccnt 0x4" ::: "memory");
}
__device__ __forceinline__ void wait_async_le0()
{
    asm volatile("s_wait_asynccnt 0x0" ::: "memory");
}

// ---------------------------------------------------------------------------
// fp32 -> bf16 convert
// ---------------------------------------------------------------------------
__global__ __launch_bounds__(256) void cvt_f32_bf16(const float* __restrict__ in,
                                                    __bf16* __restrict__ out, int n)
{
    for (int i = blockIdx.x * 256 + threadIdx.x; i < n; i += gridDim.x * 256)
        out[i] = (__bf16)in[i];
}

// ---------------------------------------------------------------------------
// Tiled fp32 -> bf16 transpose:  in [R,C] row-major  ->  out [C,R] row-major
// ---------------------------------------------------------------------------
__global__ __launch_bounds__(256) void transpose_f32_bf16(const float* __restrict__ in,
                                                          __bf16* __restrict__ out,
                                                          int R, int C)
{
    __shared__ float tile[32][33];
    const int c0 = blockIdx.x * 32, r0 = blockIdx.y * 32;
    const int tc = threadIdx.x & 31, tr = threadIdx.x >> 5;   // 8 rows / pass
    #pragma unroll
    for (int i = 0; i < 4; ++i)
        tile[tr + i * 8][tc] = in[(size_t)(r0 + tr + i * 8) * C + c0 + tc];
    __syncthreads();
    #pragma unroll
    for (int i = 0; i < 4; ++i)
        out[(size_t)(c0 + tr + i * 8) * R + r0 + tc] = (__bf16)tile[tc][tr + i * 8];
}

// ---------------------------------------------------------------------------
// 128x128 NT GEMM on bf16 WMMA:  C[m,n] = sum_k A[m,k] * B[n,k]
// 8 wave32s, 4(M) x 2(N); each wave owns 32x64: 2 A-frags x 4 B-frags
//   -> 8 v_wmma per k-step (A reused x4, B reused x2).
// 4-deep LDS ring; staging via global_load_async_to_lds_b128 (ASYNCcnt);
// steady-state body: stage ki+1 | prefetch ki+2 | s_wait_asynccnt 4 |
// one barrier | 8 WMMA.  Final k-step peeled (s_wait_asynccnt 0).
//   OUT_F32: store fp32, else bf16
//   MASK: causal (col > row -> NEG_INF); fully-masked blocks skip all math.
// ---------------------------------------------------------------------------
template<bool OUT_F32, bool MASK>
__global__ __launch_bounds__(256) void gemm_wmma_nt(
    const __bf16* __restrict__ A, int lda,
    const __bf16* __restrict__ B, int ldb,
    void* __restrict__ Cv, int ldc,
    int K)
{
    __shared__ __bf16 As[NBUF][BM * LDS_A];
    __shared__ __bf16 Bs[NBUF][BN * LDS_B];

    const int tid    = threadIdx.x;
    const int wave   = tid >> 5;
    const int wave_m = wave & 3;       // 0..3 -> 32-row slice
    const int wave_n = wave >> 2;      // 0..1 -> 64-col slice
    const int lane   = tid & 31;
    const int half   = lane >> 4;      // K-half selector (ISA bf16 layouts)
    const int ln16   = lane & 15;      // row (A) / col (B, C/D) in tile

    const int m0 = blockIdx.y * BM;
    const int n0 = blockIdx.x * BN;

    if (MASK) {
        if (n0 > m0 + BM - 1) {        // entirely above diagonal: fill only
            float* Cf = (float*)Cv;
            for (int i = tid; i < BM * BN; i += 256) {
                int r = i >> 7, c = i & 127;
                Cf[(size_t)(m0 + r) * ldc + n0 + c] = NEG_INF;
            }
            return;
        }
    }

    // per-thread staging coordinates (4 x 16B chunks per tile: 2 A + 2 B)
    const int srow0 = tid >> 2;                 // chunk tid
    const int skc0  = (tid & 3) * 8;
    const int srow1 = (tid + 256) >> 2;         // chunk tid+256
    const int skc1  = ((tid + 256) & 3) * 8;

    auto stage = [&](int ki, int buf) {
        const size_t kb = (size_t)ki * BK;
        async_copy16(&As[buf][srow0 * LDS_A + skc0],
                     &A[(size_t)(m0 + srow0) * lda + kb + skc0]);
        async_copy16(&Bs[buf][srow0 * LDS_B + skc0],
                     &B[(size_t)(n0 + srow0) * ldb + kb + skc0]);
        async_copy16(&As[buf][srow1 * LDS_A + skc1],
                     &A[(size_t)(m0 + srow1) * lda + kb + skc1]);
        async_copy16(&Bs[buf][srow1 * LDS_B + skc1],
                     &B[(size_t)(n0 + srow1) * ldb + kb + skc1]);
    };

    v8f acc[2][4];
    #pragma unroll
    for (int mi = 0; mi < 2; ++mi)
        #pragma unroll
        for (int t = 0; t < 4; ++t)
            #pragma unroll
            for (int i = 0; i < 8; ++i) acc[mi][t][i] = 0.0f;

    auto compute = [&](int buf) {
        // A fragments: 16-bit A 16x32 layout (two contiguous 16B runs per lane)
        v16bf a[2];
        #pragma unroll
        for (int mi = 0; mi < 2; ++mi) {
            const __bf16* ap =
                &As[buf][(wave_m * 32 + mi * 16 + ln16) * LDS_A + half * 8];
            v8bf lo = *(const v8bf*)ap;          // K = h*8 .. h*8+7
            v8bf hi = *(const v8bf*)(ap + 16);   // K = h*8+16 .. h*8+23
            #pragma unroll
            for (int i = 0; i < 8; ++i) { a[mi][i] = lo[i]; a[mi][i + 8] = hi[i]; }
        }
        // 8 WMMAs per k-step
        #pragma unroll
        for (int t = 0; t < 4; ++t) {
            const __bf16* bp =
                &Bs[buf][(wave_n * 64 + t * 16 + ln16) * LDS_B + half * 8];
            v8bf blo = *(const v8bf*)bp;         // K = h*8 .. h*8+7
            v8bf bhi = *(const v8bf*)(bp + 16);  // K = h*8+16 .. h*8+23
            v16bf b;
            #pragma unroll
            for (int i = 0; i < 8; ++i) { b[i] = blo[i]; b[i + 8] = bhi[i]; }
            #pragma unroll
            for (int mi = 0; mi < 2; ++mi)
                acc[mi][t] = __builtin_amdgcn_wmma_f32_16x16x32_bf16(
                    false, a[mi], false, b, (short)0, acc[mi][t], false, false);
        }
    };

    const int nk = K / BK;             // >= 2 for all our shapes
    stage(0, 0);

    // steady state: branch-free body, one barrier per k-step
    for (int ki = 0; ki < nk - 1; ++ki) {
        stage(ki + 1, (ki + 1) & (NBUF - 1));
        // gfx1250 global_prefetch_b8: warm tile ki+2 toward GL2
        __builtin_prefetch(&A[(size_t)(m0 + srow0) * lda +
                              (size_t)(ki + 2) * BK], 0, 3);
        wait_async_le4();              // tile ki landed; ki+1 in flight
        __syncthreads();
        compute(ki & (NBUF - 1));
        // no trailing barrier: tile ki+NBUF (same buffer) is issued only
        // after the barrier of iteration ki+1 orders it behind these reads
    }
    // peeled final k-step
    wait_async_le0();
    __syncthreads();
    compute((nk - 1) & (NBUF - 1));

    // ---- store (C/D: element v -> M = v + 8*half, N = ln16) ----
    #pragma unroll
    for (int mi = 0; mi < 2; ++mi)
        #pragma unroll
        for (int t = 0; t < 4; ++t)
            #pragma unroll
            for (int v = 0; v < 8; ++v) {
                int gm = m0 + wave_m * 32 + mi * 16 + half * 8 + v;
                int gn = n0 + wave_n * 64 + t * 16 + ln16;
                float val = acc[mi][t][v];
                if (MASK && gn > gm) val = NEG_INF;
                if (OUT_F32) ((float*) Cv)[(size_t)gm * ldc + gn] = val;
                else         ((__bf16*)Cv)[(size_t)gm * ldc + gn] = (__bf16)val;
            }
}

// ---------------------------------------------------------------------------
// Row softmax over S (fp32, 4096 wide); writes bf16 P *in place* at the start
// of each row's own 16KB slab (P row stride = 8192 bf16 elements).
// ---------------------------------------------------------------------------
__global__ __launch_bounds__(256) void softmax_rows(const float* __restrict__ S,
                                                    char* __restrict__ Sbytes)
{
    __shared__ float red[256];
    const int r = blockIdx.x;
    const int t = threadIdx.x;
    const float* row = S + (size_t)r * N_CTX;

    float vals[16];
    float m = NEG_INF;
    #pragma unroll
    for (int i = 0; i < 16; ++i) {
        vals[i] = row[t + i * 256];
        m = fmaxf(m, vals[i]);
    }
    red[t] = m; __syncthreads();
    for (int off = 128; off > 0; off >>= 1) {
        if (t < off) red[t] = fmaxf(red[t], red[t + off]);
        __syncthreads();
    }
    const float rowmax = red[0];
    __syncthreads();

    float s = 0.f;
    #pragma unroll
    for (int i = 0; i < 16; ++i) s += __expf(vals[i] - rowmax);
    red[t] = s; __syncthreads();
    for (int off = 128; off > 0; off >>= 1) {
        if (t < off) red[t] += red[t + off];
        __syncthreads();
    }
    const float inv = 1.0f / red[0];

    __bf16* P = (__bf16*)(Sbytes + (size_t)r * (N_CTX * sizeof(float)));
    #pragma unroll
    for (int i = 0; i < 16; ++i)
        P[t + i * 256] = (__bf16)(__expf(vals[i] - rowmax) * inv);
}

// ---------------------------------------------------------------------------
// softmax(x Wqk^T x^T + causal) x Wov^T -- all GEMMs NT on bf16 WMMA.
// Workspace layout (bytes):
//   [0,16M)     x_bf   4096x2048 bf16 (row-major)
//   [16M,32M)   q_bf   (reused as o_bf after S is built)
//   [32M,40M)   Wqk_bf
//   [40M,48M)   Wov_bf
//   [48M,112M)  S fp32 4096x4096; P bf16 in place (row stride 8192 elems)
//   [112M,128M) x_bfT  2048x4096 bf16 (x transposed, for P @ x as NT)
// ---------------------------------------------------------------------------
extern "C" void kernel_launch(void* const* d_in, const int* in_sizes, int n_in,
                              void* d_out, int out_size, void* d_ws, size_t ws_size,
                              hipStream_t stream)
{
    const float* x   = (const float*)d_in[0];
    const float* Wqk = (const float*)d_in[1];
    const float* Wov = (const float*)d_in[2];
    float* out = (float*)d_out;

    char* ws = (char*)d_ws;
    __bf16* x_bf   = (__bf16*)(ws + (size_t)0);
    __bf16* q_bf   = (__bf16*)(ws + ((size_t)16 << 20));
    __bf16* o_bf   = q_bf; // reuse after scores are computed
    __bf16* wqk_bf = (__bf16*)(ws + ((size_t)32 << 20));
    __bf16* wov_bf = (__bf16*)(ws + ((size_t)40 << 20));
    float*  S      = (float*) (ws + ((size_t)48 << 20));
    __bf16* x_bfT  = (__bf16*)(ws + ((size_t)112 << 20));

    const int nx = N_CTX * D_MODEL;    // 8M
    const int nw = D_MODEL * D_MODEL;  // 4M
    cvt_f32_bf16<<<4096, 256, 0, stream>>>(x,   x_bf,   nx);
    cvt_f32_bf16<<<2048, 256, 0, stream>>>(Wqk, wqk_bf, nw);
    cvt_f32_bf16<<<2048, 256, 0, stream>>>(Wov, wov_bf, nw);
    // x^T in bf16: [D_MODEL, N_CTX]
    transpose_f32_bf16<<<dim3(D_MODEL / 32, N_CTX / 32), 256, 0, stream>>>(
        x, x_bfT, N_CTX, D_MODEL);

    // q = x @ Wqk^T   (bf16 out)  M=4096 N=2048 K=2048
    gemm_wmma_nt<false, false><<<dim3(D_MODEL / BN, N_CTX / BM), 256, 0, stream>>>(
        x_bf, D_MODEL, wqk_bf, D_MODEL, (void*)q_bf, D_MODEL, D_MODEL);

    // S = q @ x^T + causal mask (f32 out)  M=N=4096 K=2048
    gemm_wmma_nt<true, true><<<dim3(N_CTX / BN, N_CTX / BM), 256, 0, stream>>>(
        q_bf, D_MODEL, x_bf, D_MODEL, (void*)S, N_CTX, D_MODEL);

    // P = softmax(S) rows, bf16 in place (row stride 2*N_CTX elements)
    softmax_rows<<<N_CTX, 256, 0, stream>>>(S, (char*)S);

    // o = P @ x = P @ (x^T)^T  (bf16 out)  M=4096 N=2048 K=4096
    gemm_wmma_nt<false, false><<<dim3(D_MODEL / BN, N_CTX / BM), 256, 0, stream>>>(
        (const __bf16*)S, 2 * N_CTX, x_bfT, N_CTX, (void*)o_bf, D_MODEL, N_CTX);

    // out = o @ Wov^T (f32 out)  M=4096 N=2048 K=2048
    gemm_wmma_nt<true, false><<<dim3(D_MODEL / BN, N_CTX / BM), 256, 0, stream>>>(
        o_bf, D_MODEL, wov_bf, D_MODEL, (void*)out, D_MODEL, D_MODEL);
}
